// ProbAttention_48034914239212
// MI455X (gfx1250) — matrix operand order repaired and checked by reading, not compiled, not run on hardware
//
#include <hip/hip_runtime.h>

#define DEV static __device__ __forceinline__

constexpr int   B_ = 32, H_ = 8, L_ = 4096, D_ = 64, U_ = 45, SAMP_ = 45;
constexpr int   NTILE_ = 3;                 // ceil(45/16) query tiles per (b,h)
constexpr float SCALE_ = 0.125f;            // 1/sqrt(64)
constexpr float NEG_   = -3.0e38f;

typedef __attribute__((ext_vector_type(16))) __bf16 v16bf;
typedef __attribute__((ext_vector_type(8)))  float  v8f;
typedef int v4i_ __attribute__((vector_size(16)));   // matches async-LDS builtin param type

#if defined(__has_builtin)
#if __has_builtin(__builtin_amdgcn_global_load_async_to_lds_b128)
#define ASYNC_LDS 1
#endif
#endif

DEV void wait_async0() {
#if defined(__has_builtin) && __has_builtin(__builtin_amdgcn_s_wait_asynccnt)
  __builtin_amdgcn_s_wait_asynccnt(0);
#else
  asm volatile("s_wait_asynccnt 0x0" ::: "memory");
#endif
}
DEV void wait_ds0() {
#if defined(__has_builtin) && __has_builtin(__builtin_amdgcn_s_wait_dscnt)
  __builtin_amdgcn_s_wait_dscnt(0);
#else
  asm volatile("s_wait_dscnt 0x0" ::: "memory");
#endif
}

DEV v8f wmma_bf16(v16bf a, v16bf b, v8f c) {
  return __builtin_amdgcn_wmma_f32_16x16x32_bf16(false, a, false, b, (short)0, c, false, false);
}

// native f32 -> bf16 (RNE), lets the backend use hardware cvt instead of bit twiddling
DEV void put4(v16bf& dst, int base, float4 f) {
  dst[base + 0] = (__bf16)f.x;
  dst[base + 1] = (__bf16)f.y;
  dst[base + 2] = (__bf16)f.z;
  dst[base + 3] = (__bf16)f.w;
}

// ---------------- Kernel 1: context = cumsum(V, axis=L), transposed store ----------------
__global__ __launch_bounds__(512) void cumsum_kernel(const float* __restrict__ V,
                                                     float* __restrict__ out) {
  int bh = blockIdx.x; int b = bh / H_, h = bh % H_;
  int tid = threadIdx.x;
  int d = tid & 63, seg = tid >> 6;                       // 8 segments x 512 rows
  const float* Vb = V + (size_t)bh * L_ * D_;
  __shared__ float segtot[8][64];
  __shared__ float segoff[8][64];
  int l0 = seg * 512;
  float s = 0.f;
  for (int l = l0; l < l0 + 512; ++l) s += Vb[(size_t)l * D_ + d];
  segtot[seg][d] = s;
  __syncthreads();
  if (seg == 0) {
    float run = 0.f;
#pragma unroll
    for (int s2 = 0; s2 < 8; ++s2) { float t = segtot[s2][d]; segoff[s2][d] = run; run += t; }
  }
  __syncthreads();
  float acc = segoff[seg][d];
  for (int l = l0; l < l0 + 512; ++l) {
    acc += Vb[(size_t)l * D_ + d];                        // inclusive scan (L2 re-hit)
    out[(((size_t)b * L_ + l) * H_ + h) * D_ + d] = acc;
  }
}

// ---------------- Kernel 2: sparsity measure M = max - mean of 45 sampled scores ----------
DEV float dot64(const float* a, const float* b) {
  float acc = 0.f;
  const float4* a4 = (const float4*)a; const float4* b4 = (const float4*)b;
#pragma unroll
  for (int i = 0; i < 16; ++i) {
    float4 x = a4[i], y = b4[i];
    acc += x.x * y.x + x.y * y.y + x.z * y.z + x.w * y.w;
  }
  return acc;
}

__global__ __launch_bounds__(256) void sparsity_kernel(const float* __restrict__ Q,
                                                       const float* __restrict__ K,
                                                       const int* __restrict__ idxs,
                                                       float* __restrict__ M) {
  long gq = (long)blockIdx.x * 8 + (threadIdx.x >> 5);    // one wave32 per query
  int ln = threadIdx.x & 31;
  int bh = (int)(gq >> 12); int q = (int)(gq & (L_ - 1));
  const float* qrow = Q + ((size_t)bh * L_ + q) * D_;
  const float* Kb = K + (size_t)bh * L_ * D_;
  int i0 = idxs[(size_t)q * SAMP_ + ln];
  float d0 = dot64(qrow, Kb + (size_t)i0 * D_);
  float lmax = d0, lsum = d0;
  if (ln < SAMP_ - 32) {                                  // lanes 0..12 own a 2nd sample
    int i1 = idxs[(size_t)q * SAMP_ + ln + 32];
    float d1 = dot64(qrow, Kb + (size_t)i1 * D_);
    lmax = fmaxf(lmax, d1); lsum += d1;
  }
#pragma unroll
  for (int m = 16; m >= 1; m >>= 1) {
    lmax = fmaxf(lmax, __shfl_xor(lmax, m));
    lsum += __shfl_xor(lsum, m);
  }
  if (ln == 0) M[gq] = lmax - lsum * (1.0f / (float)SAMP_);
}

// ---------------- Kernel 3: top-45 queries per (b,h) by M ----------------
__global__ __launch_bounds__(256) void topk_kernel(const float* __restrict__ M,
                                                   int* __restrict__ Mtop) {
  int bh = blockIdx.x; int tid = threadIdx.x;
  __shared__ float vals[L_];
  __shared__ float rv[256];
  __shared__ int   ri[256];
  const float* Mb = M + (size_t)bh * L_;
  for (int i = tid; i < L_; i += 256) vals[i] = Mb[i];
  __syncthreads();
  for (int it = 0; it < U_; ++it) {
    float best = NEG_; int bidx = 0;
    for (int i = tid; i < L_; i += 256) {
      float v = vals[i];
      if (v > best) { best = v; bidx = i; }
    }
    rv[tid] = best; ri[tid] = bidx;
    __syncthreads();
#pragma unroll
    for (int s = 128; s >= 1; s >>= 1) {
      if (tid < s) {
        float vo = rv[tid + s];
        if (vo > rv[tid] || (vo == rv[tid] && ri[tid + s] < ri[tid])) { rv[tid] = vo; ri[tid] = ri[tid + s]; }
      }
      __syncthreads();
    }
    if (tid == 0) { Mtop[(size_t)bh * 48 + it] = ri[0]; vals[ri[0]] = NEG_; }
    __syncthreads();
  }
}

// ---------------- Kernel 4: dense attention for the top-45 queries (flash-style, WMMA) ----
// Block = 128 threads = 4 waves. Each block handles one (b,h, 16-query tile); waves split the
// key stream in stride-128 chunks of 32 keys, each keeping an online-softmax state, merged
// at the end through LDS. V chunks are staged LDS-direct via GLOBAL_LOAD_ASYNC_TO_LDS_B128
// (ASYNCcnt), overlapping with the K loads + S-WMMAs + softmax of the same chunk.
__global__ __launch_bounds__(128) void attn_kernel(const float* __restrict__ Q,
                                                   const float* __restrict__ K,
                                                   const float* __restrict__ V,
                                                   const int* __restrict__ Mtop,
                                                   float* __restrict__ out) {
  int blk = blockIdx.x;
  int bh = blk / NTILE_, t = blk % NTILE_;
  int b = bh / H_, h = bh % H_;
  int tid = threadIdx.x;
  int wv = tid >> 5, ln = tid & 31;
  int half = ln >> 4, nn = ln & 15;

  const float* Qb = Q + (size_t)bh * L_ * D_;
  const float* Kb = K + (size_t)bh * L_ * D_;
  const float* Vb = V + (size_t)bh * L_ * D_;

  __shared__ int qpos_s[16];
  __shared__ __align__(16) float Vbuf[4][32 * D_];   // per-wave V chunk (f32)
  __shared__ __align__(16) float Pbuf[4][16 * 32];   // per-wave P tile (C/D layout -> row major)
  __shared__ float sO[4][16][D_];                    // per-wave partial O
  __shared__ float smax_s[4][16], ssum_s[4][16];
  __shared__ float wsc[4][16], ginv[16];

  if (tid < 16) {
    int row = t * 16 + tid;
    qpos_s[tid] = (row < U_) ? Mtop[(size_t)bh * 48 + row] : 0;   // pad rows -> query 0
  }
  __syncthreads();

  int   qp[8];
  float rmax[8], rsum[8];
#pragma unroll
  for (int r = 0; r < 8; ++r) { qp[r] = qpos_s[half * 8 + r]; rmax[r] = NEG_; rsum[r] = 0.f; }

  int tmax = 0;
#pragma unroll
  for (int m = 0; m < 16; ++m) tmax = max(tmax, qpos_s[m]);

  // Q A-fragments (16x32 bf16 per D-slice): lane row = nn; K interleaved by half per ISA.
  // Element runs are contiguous 8-float spans -> 2x b128 per span.
  const float* qrow = Qb + (size_t)qpos_s[nn] * D_;
  v16bf aq[2];
#pragma unroll
  for (int s = 0; s < 2; ++s) {
    const float4* qa = (const float4*)(qrow + s * 32 + 8 * half);       // K = 8h .. 8h+7
    const float4* qb4 = (const float4*)(qrow + s * 32 + 16 + 8 * half); // K = 16+8h .. 23+8h
    put4(aq[s], 0, qa[0]);  put4(aq[s], 4, qa[1]);
    put4(aq[s], 8, qb4[0]); put4(aq[s], 12, qb4[1]);
  }

  v8f o[4];
#pragma unroll
  for (int nt = 0; nt < 4; ++nt) o[nt] = (v8f){0.f, 0.f, 0.f, 0.f, 0.f, 0.f, 0.f, 0.f};

  float* vb = Vbuf[wv];
  float* pb = Pbuf[wv];

  for (int cb = wv * 32; cb <= tmax; cb += 128) {       // per-wave chunk stream
    if (cb + 128 <= tmax) {
      __builtin_prefetch(Vb + (size_t)(cb + 128 + ln) * D_, 0, 1);   // global_prefetch_b8
      __builtin_prefetch(Kb + (size_t)(cb + 128 + ln) * D_, 0, 1);
    }

    // ---- stage V[cb..cb+31][:] into per-wave LDS (lane = key row) ----
#ifdef ASYNC_LDS
    wait_ds0();   // single-buffered Vbuf: prior chunk's DS reads must retire before async writes
    {
      v4i_* g = (v4i_*)(Vb + (size_t)(cb + ln) * D_);    // global source (cast drops const)
      v4i_* l = (v4i_*)(vb + ln * D_);                   // LDS destination
#pragma unroll
      for (int i = 0; i < 16; ++i)
        __builtin_amdgcn_global_load_async_to_lds_b128(g + i, l + i, 0, 0);
    }
#else
    {
      const float4* vr = (const float4*)(Vb + (size_t)(cb + ln) * D_);
      float4* dst = (float4*)(vb + ln * D_);
#pragma unroll
      for (int i = 0; i < 16; ++i) dst[i] = vr[i];
    }
    asm volatile("" ::: "memory");
#endif

    // ---- S = Q * K^T for two 16-key subtiles, masked+scaled in C/D layout ----
    // (overlaps with the in-flight async V stage)
    float p0[8], p1[8];
#pragma unroll
    for (int sub = 0; sub < 2; ++sub) {
      int kb = cb + sub * 16;
      v8f sacc = (v8f){0.f, 0.f, 0.f, 0.f, 0.f, 0.f, 0.f, 0.f};
#pragma unroll
      for (int s = 0; s < 2; ++s) {
        // B fragment of K^T: col = key (lane nn); K-dim interleaved by half (8h.., 16+8h..)
        const float4* k4 = (const float4*)(Kb + (size_t)(kb + nn) * D_ + s * 32 + 8 * half);
        const float4* k4b = (const float4*)(Kb + (size_t)(kb + nn) * D_ + s * 32 + 16 + 8 * half);
        v16bf bk;
        put4(bk, 0, k4[0]); put4(bk, 4, k4[1]); put4(bk, 8, k4b[0]); put4(bk, 12, k4b[1]);
        sacc = wmma_bf16(aq[s], bk, sacc);
      }
      int key = kb + nn;                                 // C/D: this lane's column
      float* pd = sub ? p1 : p0;
#pragma unroll
      for (int r = 0; r < 8; ++r) {                      // C/D: row = 8*half + r
        float sc = sacc[r] * SCALE_;
        pd[r] = (key > qp[r]) ? NEG_ : sc;               // causal mask at gathered q pos
      }
    }

    // ---- online softmax update (row reductions = shfl within 16-lane half) ----
    float nmax[8];
#pragma unroll
    for (int r = 0; r < 8; ++r) nmax[r] = fmaxf(p0[r], p1[r]);
#pragma unroll
    for (int m = 8; m >= 1; m >>= 1)
#pragma unroll
      for (int r = 0; r < 8; ++r) nmax[r] = fmaxf(nmax[r], __shfl_xor(nmax[r], m));

    float lsum[8];
#pragma unroll
    for (int r = 0; r < 8; ++r) {
      float nm = fmaxf(rmax[r], nmax[r]);
      float cr = __expf(rmax[r] - nm);
      rmax[r] = nm;
      p0[r] = __expf(p0[r] - nm);
      p1[r] = __expf(p1[r] - nm);
      lsum[r] = p0[r] + p1[r];
      rsum[r] *= cr;
#pragma unroll
      for (int nt = 0; nt < 4; ++nt) o[nt][r] *= cr;
    }
#pragma unroll
    for (int m = 8; m >= 1; m >>= 1)
#pragma unroll
      for (int r = 0; r < 8; ++r) lsum[r] += __shfl_xor(lsum[r], m);
#pragma unroll
    for (int r = 0; r < 8; ++r) rsum[r] += lsum[r];

    // ---- transpose P from C/D layout to A layout via LDS ----
#pragma unroll
    for (int r = 0; r < 8; ++r) {
      int row = half * 8 + r;
      pb[row * 32 + nn]      = p0[r];
      pb[row * 32 + 16 + nn] = p1[r];
    }
    asm volatile("" ::: "memory");

    v16bf ap;                                            // A fragment: row = nn, 32 keys
    {
      const float4* pa = (const float4*)(pb + nn * 32 + 8 * half);
      const float4* pbb = (const float4*)(pb + nn * 32 + 16 + 8 * half);
      put4(ap, 0, pa[0]);  put4(ap, 4, pa[1]);
      put4(ap, 8, pbb[0]); put4(ap, 12, pbb[1]);
    }

#ifdef ASYNC_LDS
    wait_async0();                                       // V chunk resident in LDS
#endif

    // ---- O += P @ Vchunk (4 d-tiles of 16) ----
#pragma unroll
    for (int nt = 0; nt < 4; ++nt) {
      v16bf bv;                                          // B fragment: col = d (lane nn)
#pragma unroll
      for (int v = 0; v < 8; ++v) {
        int k = (v < 4) ? 8 * half + 2 * v : 16 + 8 * half + 2 * (v - 4);
        bv[2 * v]     = (__bf16)vb[k * D_ + nt * 16 + nn];
        bv[2 * v + 1] = (__bf16)vb[(k + 1) * D_ + nt * 16 + nn];
      }
      o[nt] = wmma_bf16(ap, bv, o[nt]);
    }
    asm volatile("" ::: "memory");
  }

  // ---- publish per-wave partial state ----
  if (nn == 0) {
#pragma unroll
    for (int r = 0; r < 8; ++r) {
      smax_s[wv][half * 8 + r] = rmax[r];
      ssum_s[wv][half * 8 + r] = rsum[r];
    }
  }
#pragma unroll
  for (int nt = 0; nt < 4; ++nt)
#pragma unroll
    for (int r = 0; r < 8; ++r) sO[wv][half * 8 + r][nt * 16 + nn] = o[nt][r];
  __syncthreads();

  // ---- merge 4 wave states: O = sum_w O_w*exp(m_w-g) / sum_w s_w*exp(m_w-g) ----
  if (tid < 16) {
    float g = smax_s[0][tid];
#pragma unroll
    for (int w = 1; w < 4; ++w) g = fmaxf(g, smax_s[w][tid]);
    float stot = 0.f;
#pragma unroll
    for (int w = 0; w < 4; ++w) {
      float e = __expf(smax_s[w][tid] - g);
      wsc[w][tid] = e;
      stot += ssum_s[w][tid] * e;
    }
    ginv[tid] = 1.0f / stot;
  }
  __syncthreads();

  for (int i = tid; i < 16 * D_; i += 128) {
    int m = i >> 6, d = i & (D_ - 1);
    int rowg = t * 16 + m;
    if (rowg >= U_) continue;                            // skip pad rows
    float val = 0.f;
#pragma unroll
    for (int w = 0; w < 4; ++w) val += sO[w][m][d] * wsc[w][m];
    out[(((size_t)b * L_ + qpos_s[m]) * H_ + h) * D_ + d] = val * ginv[m];
  }
}

// ---------------- host launch ----------------
extern "C" void kernel_launch(void* const* d_in, const int* in_sizes, int n_in,
                              void* d_out, int out_size, void* d_ws, size_t ws_size,
                              hipStream_t stream) {
  (void)in_sizes; (void)n_in; (void)out_size; (void)ws_size;
  const float* Q    = (const float*)d_in[0];
  const float* K    = (const float*)d_in[1];
  const float* V    = (const float*)d_in[2];
  const int*   idxs = (const int*)d_in[3];
  // d_in[4] = attn_mask presence flag (unused)
  float* out = (float*)d_out;

  float* Mbuf = (float*)d_ws;                                        // B*H*L f32 = 4 MB
  int*   Mtop = (int*)((char*)d_ws + (size_t)B_ * H_ * L_ * sizeof(float)); // B*H*48 i32

  cumsum_kernel  <<<B_ * H_,            512, 0, stream>>>(V, out);
  sparsity_kernel<<<(B_ * H_ * L_) / 8, 256, 0, stream>>>(Q, K, idxs, Mbuf);
  topk_kernel    <<<B_ * H_,            256, 0, stream>>>(Mbuf, Mtop);
  attn_kernel    <<<B_ * H_ * NTILE_,   128, 0, stream>>>(Q, K, V, Mtop, out);
}
